// MLASegmentedCrossAttention_36043365548098
// MI455X (gfx1250) — matrix operand order repaired and checked by reading, not compiled, not run on hardware
//
#include <hip/hip_runtime.h>

typedef __attribute__((ext_vector_type(16))) _Float16 v16h;
typedef __attribute__((ext_vector_type(8)))  _Float16 v8h;
typedef __attribute__((ext_vector_type(8)))  float    v8f;

// Problem constants (from reference)
#define LQ   2048
#define LK   2048
#define QD   1024
#define KVD  1024
#define CC   256
#define QC   512
#define NH   16
#define DH   64
#define DR   64

// ---------------------------------------------------------------------------
// f32 -> f16 elementwise convert
// ---------------------------------------------------------------------------
__global__ __launch_bounds__(256) void cvt_f16(const float* __restrict__ in,
                                               _Float16* __restrict__ out, int n) {
  int i = blockIdx.x * 256 + threadIdx.x;
  if (i < n) out[i] = (_Float16)in[i];
}

// ---------------------------------------------------------------------------
// Transpose + convert: in = K x N (f32, row major) -> out = N x K (f16)
// LDS-tiled for coalescing on both sides. K, N multiples of 32.
// ---------------------------------------------------------------------------
__global__ __launch_bounds__(256) void transpose_cvt(const float* __restrict__ in,
                                                     _Float16* __restrict__ out,
                                                     int K, int N) {
  __shared__ _Float16 tile[32][33];
  const int bk = blockIdx.y * 32, bn = blockIdx.x * 32;
  const int tx = threadIdx.x & 31, ty = threadIdx.x >> 5;  // 32 x 8
#pragma unroll
  for (int i = 0; i < 32; i += 8)
    tile[ty + i][tx] = (_Float16)in[(size_t)(bk + ty + i) * N + (bn + tx)];
  __syncthreads();
#pragma unroll
  for (int i = 0; i < 32; i += 8)
    out[(size_t)(bn + ty + i) * K + (bk + tx)] = tile[tx][ty + i];
}

// ---------------------------------------------------------------------------
// WMMA GEMM: C[M,N] = A[M,K] * B[K,N]
//   A  : M x K  f16 row-major
//   BT : N x K  f16 (B transposed) -> contiguous K per output column
//   Cf : optional f32 out, Ch : optional f16 out
// Template: CT = 16-wide column tiles per wave (wave tile = 32 x 16*CT)
//           WM x WN = wave grid per block (block tile = 32*WM x 16*CT*WN)
// Main config <4,4,2>: 256 thr, block 128x128, 8 WMMA / 32-k step / wave.
// Small config <2,2,2>: 128 thr, block 64x64 (for N=64).
// Ping-pong pipelined over two named fragment sets (no register rotation).
// M multiple of 32*WM; N multiple of 16*CT*WN; K multiple of 64 and >= 64.
// ---------------------------------------------------------------------------
template <int CT, int WM, int WN>
__global__ __launch_bounds__(WM * WN * 32) void gemm_wmma_f16(
    const _Float16* __restrict__ A, const _Float16* __restrict__ BT,
    float* __restrict__ Cf, _Float16* __restrict__ Ch, int M, int N, int K) {
  const int lane = threadIdx.x & 31;
  const int wave = threadIdx.x >> 5;
  const int lm = lane & 15;       // row (A) / col (B) within 16
  const int lh = lane >> 4;       // half-wave selector
  const int tileM = blockIdx.y * (32 * WM) + (wave / WN) * 32;
  const int tileN = blockIdx.x * (16 * CT * WN) + (wave % WN) * (16 * CT);

  v8f acc0[CT], acc1[CT];
#pragma unroll
  for (int t = 0; t < CT; ++t) { acc0[t] = (v8f){}; acc1[t] = (v8f){}; }

  // A-fragment (16-bit A 16x32 layout): lane holds row m=lm, K = lh*16..+15 (contiguous, matches B)
  const _Float16* a0p = A + (size_t)(tileM + lm) * K + lh * 16;
  const _Float16* a1p = A + (size_t)(tileM + 16 + lm) * K + lh * 16;
  // B-fragment (16-bit B 32x16 layout): lane holds col n=lm, K = lh*16..+15 (contiguous in BT)
  const _Float16* bp[CT];
#pragma unroll
  for (int t = 0; t < CT; ++t)
    bp[t] = BT + (size_t)(tileN + 16 * t + lm) * K + lh * 16;

  auto load_frags = [&](int k0, v16h& A0, v16h& A1, v16h (&B)[CT]) {
    v8h a0lo = *(const v8h*)(a0p + k0);
    v8h a0hi = *(const v8h*)(a0p + k0 + 8);
    v8h a1lo = *(const v8h*)(a1p + k0);
    v8h a1hi = *(const v8h*)(a1p + k0 + 8);
#pragma unroll
    for (int t = 0; t < CT; ++t) B[t] = *(const v16h*)(bp[t] + k0);
#pragma unroll
    for (int i = 0; i < 8; ++i) {
      A0[i] = a0lo[i]; A0[8 + i] = a0hi[i];
      A1[i] = a1lo[i]; A1[8 + i] = a1hi[i];
    }
  };

  auto do_wmma = [&](const v16h& A0, const v16h& A1, const v16h (&B)[CT]) {
#pragma unroll
    for (int t = 0; t < CT; ++t) {
      acc0[t] = __builtin_amdgcn_wmma_f32_16x16x32_f16(false, A0, false, B[t], (short)0, acc0[t], false, false);
      acc1[t] = __builtin_amdgcn_wmma_f32_16x16x32_f16(false, A1, false, B[t], (short)0, acc1[t], false, false);
    }
  };

  // Ping-pong software pipeline (K multiple of 64, K >= 64).
  v16h xa0, xa1, xb[CT];   // set A
  v16h ya0, ya1, yb[CT];   // set B
  load_frags(0, xa0, xa1, xb);
  int k0 = 0;
  for (; k0 + 64 < K; k0 += 64) {
    load_frags(k0 + 32, ya0, ya1, yb);   // prefetch odd step
    do_wmma(xa0, xa1, xb);               // compute even step
    load_frags(k0 + 64, xa0, xa1, xb);   // prefetch next even step
    do_wmma(ya0, ya1, yb);               // compute odd step
  }
  // Epilogue: last 64 of K
  load_frags(k0 + 32, ya0, ya1, yb);
  do_wmma(xa0, xa1, xb);
  do_wmma(ya0, ya1, yb);

  // C/D layout: VGPR j -> row j + 8*lh; lane lm -> column
#pragma unroll
  for (int t = 0; t < CT; ++t) {
    const int col = tileN + 16 * t + lm;
#pragma unroll
    for (int j = 0; j < 8; ++j) {
      const int r0 = tileM + lh * 8 + j;
      const int r1 = r0 + 16;
      if (Cf) {
        Cf[(size_t)r0 * N + col] = acc0[t][j];
        Cf[(size_t)r1 * N + col] = acc1[t][j];
      }
      if (Ch) {
        Ch[(size_t)r0 * N + col] = (_Float16)acc0[t][j];
        Ch[(size_t)r1 * N + col] = (_Float16)acc1[t][j];
      }
    }
  }
}

// ---------------------------------------------------------------------------
// RoPE in-place on f16 buffer of shape [rows, 64]; pos = row / posdiv
// ---------------------------------------------------------------------------
__global__ __launch_bounds__(256) void rope_f16(_Float16* __restrict__ buf,
                                                int rows, int posdiv) {
  int idx = blockIdx.x * 256 + threadIdx.x;
  if (idx >= rows * 32) return;
  const int r = idx >> 5;
  const int i = idx & 31;
  const float pos = (float)(r / posdiv);
  const float inv = __powf(10000.0f, -(float)(2 * i) / 64.0f);
  float s, c;
  __sincosf(pos * inv, &s, &c);
  const float x1 = (float)buf[r * 64 + i];
  const float x2 = (float)buf[r * 64 + i + 32];
  buf[r * 64 + i]      = (_Float16)(x1 * c - x2 * s);
  buf[r * 64 + i + 32] = (_Float16)(x2 * c + x1 * s);
}

// ---------------------------------------------------------------------------
// Sparse segmented attention: each query q attends to keys seg[q]-2..seg[q].
// One wave per (q, h); lane handles 2 of the 64 output dims.
// ---------------------------------------------------------------------------
__global__ __launch_bounds__(128) void attn_sparse(
    const int* __restrict__ seg,
    const _Float16* __restrict__ qn,  // [LQ, NH*DH]
    const _Float16* __restrict__ qp,  // [LQ, NH*DR]
    const _Float16* __restrict__ kn,  // [LK, NH*DH]
    const _Float16* __restrict__ kp,  // [LK, DR]
    const _Float16* __restrict__ vv,  // [LK, NH*DH]
    _Float16* __restrict__ out) {     // [LQ, NH*DH]
  const int q    = blockIdx.y;
  const int h    = blockIdx.x * 4 + (threadIdx.x >> 5);
  const int lane = threadIdx.x & 31;
  const int d0 = lane * 2, d1 = d0 + 1;
  const int base = q * (NH * DH) + h * DH;
  const int s = seg[q];
  const float scale = 0.088388347648318447f;  // 1/sqrt(DH+DR)

  const float qn0 = (float)qn[base + d0], qn1 = (float)qn[base + d1];
  const float qp0 = (float)qp[base + d0], qp1 = (float)qp[base + d1];

  float sc[3];
#pragma unroll
  for (int t = 0; t < 3; ++t) {
    const int k = s - 2 + t;           // wave-uniform
    float v = -1e30f;
    if (k >= 0) {
      const int kb = k * (NH * DH) + h * DH;
      float partial = qn0 * (float)kn[kb + d0] + qn1 * (float)kn[kb + d1] +
                      qp0 * (float)kp[k * DR + d0] + qp1 * (float)kp[k * DR + d1];
#pragma unroll
      for (int off = 16; off > 0; off >>= 1)
        partial += __shfl_xor(partial, off, 32);
      v = partial * scale;
    }
    sc[t] = v;
  }
  const float m = fmaxf(sc[0], fmaxf(sc[1], sc[2]));
  float p[3];
  p[0] = (s - 2 >= 0) ? __expf(sc[0] - m) : 0.0f;
  p[1] = (s - 1 >= 0) ? __expf(sc[1] - m) : 0.0f;
  p[2] = __expf(sc[2] - m);
  const float denom = p[0] + p[1] + p[2];

  float o0 = 0.0f, o1 = 0.0f;
#pragma unroll
  for (int t = 0; t < 3; ++t) {
    const int k = s - 2 + t;
    if (k >= 0) {
      const float w = p[t] / denom;
      const int kb = k * (NH * DH) + h * DH;
      o0 += w * (float)vv[kb + d0];
      o1 += w * (float)vv[kb + d1];
    }
  }
  out[base + d0] = (_Float16)o0;
  out[base + d1] = (_Float16)o1;
}

// ---------------------------------------------------------------------------
extern "C" void kernel_launch(void* const* d_in, const int* in_sizes, int n_in,
                              void* d_out, int out_size, void* d_ws, size_t ws_size,
                              hipStream_t stream) {
  const float* q_f32  = (const float*)d_in[0];
  const float* kv_f32 = (const float*)d_in[1];
  const float* W_kv = (const float*)d_in[2];
  const float* W_dq = (const float*)d_in[3];
  const float* W_uq = (const float*)d_in[4];
  const float* W_qr = (const float*)d_in[5];
  const float* W_uk = (const float*)d_in[6];
  const float* W_kr = (const float*)d_in[7];
  const float* W_uv = (const float*)d_in[8];
  const float* W_o  = (const float*)d_in[9];
  const int*   seg  = (const int*)d_in[10];
  float* out_f32 = (float*)d_out;

  size_t off = 0;
  auto alloc = [&](size_t elems) -> _Float16* {
    _Float16* p = (_Float16*)((char*)d_ws + off);
    off = (off + elems * sizeof(_Float16) + 255) & ~(size_t)255;
    return p;
  };

  _Float16* q16    = alloc((size_t)LQ * QD);
  _Float16* kv16   = alloc((size_t)LK * KVD);
  _Float16* WkvT   = alloc((size_t)CC * KVD);        // [256][1024]
  _Float16* WdqT   = alloc((size_t)QC * QD);         // [512][1024]
  _Float16* WuqT   = alloc((size_t)(NH * DH) * QC);  // [1024][512]
  _Float16* WqrT   = alloc((size_t)(NH * DR) * QC);  // [1024][512]
  _Float16* WukT   = alloc((size_t)(NH * DH) * CC);  // [1024][256]
  _Float16* WkrT   = alloc((size_t)DR * CC);         // [64][256]
  _Float16* WuvT   = alloc((size_t)(NH * DH) * CC);  // [1024][256]
  _Float16* WoT    = alloc((size_t)QD * (NH * DH));  // [1024][1024]
  _Float16* kvc16  = alloc((size_t)LK * CC);
  _Float16* qc16   = alloc((size_t)LQ * QC);
  _Float16* qn16   = alloc((size_t)LQ * NH * DH);
  _Float16* qp16   = alloc((size_t)LQ * NH * DR);
  _Float16* kn16   = alloc((size_t)LK * NH * DH);
  _Float16* kp16   = alloc((size_t)LK * DR);
  _Float16* v16    = alloc((size_t)LK * NH * DH);
  _Float16* out16  = alloc((size_t)LQ * NH * DH);
  (void)ws_size; (void)n_in; (void)in_sizes; (void)out_size;

  // 1) Activations f32 -> f16
  cvt_f16<<<(LQ * QD + 255) / 256, 256, 0, stream>>>(q_f32, q16, LQ * QD);
  cvt_f16<<<(LK * KVD + 255) / 256, 256, 0, stream>>>(kv_f32, kv16, LK * KVD);

  // 2) Weights: transpose + convert ([K][N] f32 -> [N][K] f16)
  transpose_cvt<<<dim3(CC / 32, KVD / 32), 256, 0, stream>>>(W_kv, WkvT, KVD, CC);
  transpose_cvt<<<dim3(QC / 32, QD / 32), 256, 0, stream>>>(W_dq, WdqT, QD, QC);
  transpose_cvt<<<dim3((NH * DH) / 32, QC / 32), 256, 0, stream>>>(W_uq, WuqT, QC, NH * DH);
  transpose_cvt<<<dim3((NH * DR) / 32, QC / 32), 256, 0, stream>>>(W_qr, WqrT, QC, NH * DR);
  transpose_cvt<<<dim3((NH * DH) / 32, CC / 32), 256, 0, stream>>>(W_uk, WukT, CC, NH * DH);
  transpose_cvt<<<dim3(DR / 32, CC / 32), 256, 0, stream>>>(W_kr, WkrT, CC, DR);
  transpose_cvt<<<dim3((NH * DH) / 32, CC / 32), 256, 0, stream>>>(W_uv, WuvT, CC, NH * DH);
  transpose_cvt<<<dim3(QD / 32, (NH * DH) / 32), 256, 0, stream>>>(W_o, WoT, NH * DH, QD);

  // 3) Projection GEMMs: block tile 128x128, wave tile 32x64, 8 WMMA/k-step
  gemm_wmma_f16<4, 4, 2><<<dim3(CC / 128, LK / 128), 256, 0, stream>>>(kv16, WkvT, nullptr, kvc16, LK, CC, KVD);
  gemm_wmma_f16<4, 4, 2><<<dim3(QC / 128, LQ / 128), 256, 0, stream>>>(q16, WdqT, nullptr, qc16, LQ, QC, QD);
  gemm_wmma_f16<4, 4, 2><<<dim3((NH * DH) / 128, LQ / 128), 256, 0, stream>>>(qc16, WuqT, nullptr, qn16, LQ, NH * DH, QC);
  gemm_wmma_f16<4, 4, 2><<<dim3((NH * DR) / 128, LQ / 128), 256, 0, stream>>>(qc16, WqrT, nullptr, qp16, LQ, NH * DR, QC);
  gemm_wmma_f16<4, 4, 2><<<dim3((NH * DH) / 128, LK / 128), 256, 0, stream>>>(kvc16, WukT, nullptr, kn16, LK, NH * DH, CC);
  gemm_wmma_f16<2, 2, 2><<<dim3(DR / 64, LK / 64), 128, 0, stream>>>(kvc16, WkrT, nullptr, kp16, LK, DR, CC);
  gemm_wmma_f16<4, 4, 2><<<dim3((NH * DH) / 128, LK / 128), 256, 0, stream>>>(kvc16, WuvT, nullptr, v16, LK, NH * DH, CC);

  // 4) RoPE (pos = arange per reference)
  rope_f16<<<(LQ * NH * 32 + 255) / 256, 256, 0, stream>>>(qp16, LQ * NH, NH);
  rope_f16<<<(LK * 32 + 255) / 256, 256, 0, stream>>>(kp16, LK, 1);

  // 5) Sparse attention (<=3 keys per query)
  attn_sparse<<<dim3(NH / 4, LQ), 128, 0, stream>>>(seg, qn16, qp16, kn16, kp16, v16, out16);

  // 6) Output projection -> f32 d_out
  gemm_wmma_f16<4, 4, 2><<<dim3(QD / 128, LQ / 128), 256, 0, stream>>>(out16, WoT, out_f32, nullptr, LQ, QD, NH * DH);
}
